// DAWN_Gumble_15083925143621
// MI455X (gfx1250) — compile-verified, matrix operand order checked by reading, not run on hardware
//
#include <hip/hip_runtime.h>
#include <hip/hip_bf16.h>

typedef __attribute__((ext_vector_type(16))) _Float16 v16h;
typedef __attribute__((ext_vector_type(8)))  float    v8f;

#define NB 128   // batch
#define NC 16    // channels == WMMA tile
#define NH 9
#define NT 2048
#define NL 9
#define TT 128   // time tile per block (8 waves x 16 columns)

__device__ __forceinline__ int kOf(int hi, int e) {
    // K index held by (lane-half hi, vector element e) for 16-bit A/B fragments
    return (e < 8 ? e : e + 8) + (hi ? 8 : 0);
}

__device__ __forceinline__ v8f wmma16(v16h a, v16h b, v8f c) {
    return __builtin_amdgcn_wmma_f32_16x16x32_f16(false, a, false, b, (short)0, c, false, false);
}

__device__ __forceinline__ float prelu(float v, float a) { return v >= 0.0f ? v : a * v; }

// -------- Fused front-end: conv1(33) + bias + BN + PReLU + conv2(3, WMMA) + bias + BN + PReLU
__global__ __launch_bounds__(256) void front_kernel(
    const float* __restrict__ x,
    const float* __restrict__ Wc1, const float* __restrict__ bc1,
    const float* __restrict__ bn1g, const float* __restrict__ bn1b, const float* __restrict__ a1p,
    const float* __restrict__ Wc2, const float* __restrict__ bc2,
    const float* __restrict__ bn2g, const float* __restrict__ bn2b, const float* __restrict__ a2p,
    float* __restrict__ y2)
{
    __shared__ float    lds_x[TT + 36];          // x[T0-17 .. T0+144]
    __shared__ _Float16 lds_a1[NC][TT + 4];      // post conv1+bn+prelu, tt in [0, TT+1]
    const int bh = blockIdx.x, b = bh / NH, h = bh % NH;
    const int T0 = blockIdx.y * TT;
    const int tid = threadIdx.x;
    const float a_c1 = a1p[0], a_c2 = a2p[0];

    const float* xrow = x + (size_t)(b * NH + h) * NT;
    for (int j = tid; j < TT + 34; j += 256) {
        int t = T0 - 17 + j;
        lds_x[j] = (t >= 0 && t < NT) ? xrow[t] : 0.0f;   // conv1 zero pad
    }
    __syncthreads();

    for (int idx = tid; idx < NC * (TT + 2); idx += 256) {
        int c  = idx / (TT + 2);
        int tt = idx % (TT + 2);
        int t1 = T0 - 1 + tt;
        _Float16 out = (_Float16)0.0f;
        if (t1 >= 0 && t1 < NT) {                // conv2 zero pad: OOB y1 contributes 0
            float acc = bc1[c];
            const float* w = Wc1 + c * 33;
            #pragma unroll
            for (int k = 0; k < 33; ++k) acc += lds_x[tt + k] * w[k];
            out = (_Float16)prelu(acc * bn1g[c] + bn1b[c], a_c1);
        }
        lds_a1[c][tt] = out;
    }
    __syncthreads();

    const int wave = tid >> 5, lane = tid & 31, n = lane & 15, hi = lane >> 4;
    const int lb = wave * 16;
    const int m = n;                              // A-fragment row
    v16h aA, aB, bA, bB;
    #pragma unroll
    for (int e = 0; e < 16; ++e) {
        int K = kOf(hi, e), tap = K >> 4, cin = K & 15;
        aA[e] = (_Float16)Wc2[(m * NC + cin) * 3 + tap];                     // taps 0,1
        aB[e] = (K < 16) ? (_Float16)Wc2[(m * NC + K) * 3 + 2] : (_Float16)0.0f; // tap 2
        bA[e] = lds_a1[cin][lb + n + tap];        // offset -1 / 0
        bB[e] = (K < 16) ? lds_a1[K][lb + n + 2] : (_Float16)0.0f;           // offset +1
    }
    v8f acc = {};
    acc = wmma16(aA, bA, acc);
    acc = wmma16(aB, bB, acc);

    const int tq = T0 + lb + n;
    #pragma unroll
    for (int r = 0; r < 8; ++r) {
        int mo = r + 8 * hi;
        float v = acc[r] + bc2[mo];
        v = prelu(v * bn2g[mo] + bn2b[mo], a_c2);
        y2[((size_t)(b * NC + mo) * NH + h) * NT + tq] = v;
    }
}

// -------- Lifting level: mode 0 -> c = s*even + tanh(W2*prelu(W1*reflpad(s*odd)))  (store c, sum_c)
//                         mode 1 -> d = s*odd  - tanh(W2*prelu(W1*reflpad(c)))      (sum_d only)
__global__ __launch_bounds__(256) void lift_kernel(
    const float* __restrict__ xprev,   // (B,C,H,2*tL)
    const float* __restrict__ zsrc,    // mode1: c buffer (B,C,H,tL)
    int tL,
    const float* __restrict__ W1, const float* __restrict__ W2, const float* __restrict__ aPtr,
    const float* __restrict__ scale, int use_scale, int mode,
    float* __restrict__ cbuf, float* __restrict__ sums)
{
    __shared__ _Float16 lz[NC][TT + 4];   // z[T0-1 .. T0+TT] reflect-mapped
    const int bh = blockIdx.x, b = bh / NH, h = bh % NH;
    const int T0 = blockIdx.y * TT;
    const int tid = threadIdx.x;
    const float s  = use_scale ? scale[b] : 1.0f;
    const float al = aPtr[0];

    for (int idx = tid; idx < NC * (TT + 2); idx += 256) {
        int cin = idx / (TT + 2);
        int tt  = idx % (TT + 2);
        int ti  = T0 - 1 + tt;
        if (ti < 0) ti = -ti;                       // reflect pad
        if (ti >= tL) ti = 2 * tL - 2 - ti;
        ti = max(0, min(tL - 1, ti));               // clamp for far-OOB garbage columns
        float zv;
        if (mode == 0) zv = s * xprev[((size_t)(b * NC + cin) * NH + h) * (2 * tL) + 2 * ti + 1];
        else           zv =     zsrc [((size_t)(b * NC + cin) * NH + h) * tL + ti];
        lz[cin][tt] = (_Float16)zv;
    }
    __syncthreads();

    const int wave = tid >> 5, lane = tid & 31, n = lane & 15, hi = lane >> 4;
    const int lb = wave * 16;
    if (T0 + lb >= tL) return;                      // wave-uniform exit (no barriers below)

    const int m = n;
    v16h aA, aB, aC, bA, bB;
    #pragma unroll
    for (int e = 0; e < 16; ++e) {
        int K = kOf(hi, e), tap = K >> 4, cin = K & 15;
        aA[e] = (_Float16)W1[(m * NC + cin) * 3 + tap];
        aB[e] = (K < 16) ? (_Float16)W1[(m * NC + K) * 3 + 2] : (_Float16)0.0f;
        aC[e] = (K < 16) ? (_Float16)W2[m * NC + K]           : (_Float16)0.0f;
        bA[e] = lz[cin][lb + n + tap];
        bB[e] = (K < 16) ? lz[K][lb + n + 2] : (_Float16)0.0f;
    }
    v8f acc = {};
    acc = wmma16(aA, bA, acc);
    acc = wmma16(aB, bB, acc);

    // PReLU; C/D layout coincides with B layout for K<16 -> feed accumulator straight back in-lane
    v16h bC;
    #pragma unroll
    for (int e = 0; e < 16; ++e)
        bC[e] = (e < 8) ? (_Float16)prelu(acc[e], al) : (_Float16)0.0f;
    v8f acc2 = {};
    acc2 = wmma16(aC, bC, acc2);

    const int tq = T0 + lb + n;
    const bool valid = (tq < tL);
    #pragma unroll
    for (int r = 0; r < 8; ++r) {
        int mo = r + 8 * hi;
        float br = tanhf(acc2[r]);
        float outv = 0.0f;
        if (mode == 0) {
            float ev = valid ? s * xprev[((size_t)(b * NC + mo) * NH + h) * (2 * tL) + 2 * tq] : 0.0f;
            outv = ev + br;
            if (valid) cbuf[((size_t)(b * NC + mo) * NH + h) * tL + tq] = outv;
        } else {
            float ov = valid ? s * xprev[((size_t)(b * NC + mo) * NH + h) * (2 * tL) + 2 * tq + 1] : 0.0f;
            outv = ov - br;
        }
        float v = valid ? outv : 0.0f;
        #pragma unroll
        for (int off = 1; off < 16; off <<= 1) v += __shfl_xor(v, off, 32);  // 16-lane half reduce
        if (n == 0) atomicAdd(&sums[((size_t)(b * 2 + mode) * NC + mo) * NH + h], v);
    }
}

// -------- Gate: pooled means -> 2 matvecs -> softmax -> +gumbel -> softmax -> m, next-level scale
__global__ __launch_bounds__(32) void gate_kernel(
    const float* __restrict__ sums, const float* __restrict__ Wg1, const float* __restrict__ Wg2,
    const float* __restrict__ ag1, const float* __restrict__ ag2, const float* __restrict__ gum,
    int lvl, int tL, float* __restrict__ mbuf, float* __restrict__ scl)
{
    __shared__ float pooled[2][NC];
    __shared__ float hm1[2][NC];
    const int b = blockIdx.x, tid = threadIdx.x;
    const int p = tid >> 4, c = tid & 15;
    float acc = 0.0f;
    for (int h = 0; h < NH; ++h) acc += sums[((size_t)(b * 2 + p) * NC + c) * NH + h];
    pooled[p][c] = acc / (float)(NH * tL);
    __syncthreads();
    const float* w1 = Wg1 + lvl * NC * NC;
    float a1 = ag1[lvl], d = 0.0f;
    for (int k = 0; k < NC; ++k) d += w1[c * NC + k] * pooled[p][k];
    hm1[p][c] = prelu(d, a1);
    __syncthreads();
    if (c == 0) {
        const float* w2 = Wg2 + lvl * 2 * NC;
        float a2 = ag2[lvl], h0 = 0.0f, h1 = 0.0f;
        for (int k = 0; k < NC; ++k) { h0 += w2[k] * hm1[p][k]; h1 += w2[NC + k] * hm1[p][k]; }
        h0 = prelu(h0, a2); h1 = prelu(h1, a2);
        float mx = fmaxf(h0, h1);
        float e0 = expf(h0 - mx), e1 = expf(h1 - mx), inv = 1.0f / (e0 + e1);
        float s0 = e0 * inv, s1 = e1 * inv;
        const float* g = gum + ((size_t)lvl * 2 * NB + (b * 2 + p)) * 2;
        float u0 = s0 + g[0], u1 = s1 + g[1];                 // TAU = 1
        float mx2 = fmaxf(u0, u1);
        float f0 = expf(u0 - mx2), f1 = expf(u1 - mx2), inv2 = 1.0f / (f0 + f1);
        mbuf[(b * 2 + p) * 2 + 0] = f0 * inv2;
        mbuf[(b * 2 + p) * 2 + 1] = f1 * inv2;
        if (p == 0) scl[b] = f0 * inv2;                        // m[b,0,0] scales next level
    }
}

// -------- Output slice: feats[b,p,c,h,lvl] = m[b,p,1] * mean_t(out)
__global__ void feats_kernel(const float* __restrict__ sums, const float* __restrict__ mbuf,
                             int lvl, int tL, float* __restrict__ out)
{
    int idx = blockIdx.x * blockDim.x + threadIdx.x;
    if (idx >= NB * 2 * NC * NH) return;
    int h = idx % NH, t = idx / NH;
    int c = t % NC;  t /= NC;
    int p = t & 1;   int b = t >> 1;
    float m1 = mbuf[(b * 2 + p) * 2 + 1];
    out[(((size_t)(b * 2 + p) * NC + c) * NH + h) * NL + lvl] = m1 * sums[idx] / (float)tL;
}

extern "C" void kernel_launch(void* const* d_in, const int* in_sizes, int n_in,
                              void* d_out, int out_size, void* d_ws, size_t ws_size,
                              hipStream_t stream) {
    (void)in_sizes; (void)n_in; (void)out_size; (void)ws_size;
    const float* x    = (const float*)d_in[0];
    const float* gum  = (const float*)d_in[1];
    const float* Wc1  = (const float*)d_in[2];
    const float* bc1  = (const float*)d_in[3];
    const float* bn1g = (const float*)d_in[4];
    const float* bn1b = (const float*)d_in[5];
    const float* ac1  = (const float*)d_in[6];
    const float* Wc2  = (const float*)d_in[7];
    const float* bc2  = (const float*)d_in[8];
    const float* bn2g = (const float*)d_in[9];
    const float* bn2b = (const float*)d_in[10];
    const float* ac2  = (const float*)d_in[11];
    const float* WP1  = (const float*)d_in[12];
    const float* aP   = (const float*)d_in[13];
    const float* WP2  = (const float*)d_in[14];
    const float* WU1  = (const float*)d_in[15];
    const float* aU   = (const float*)d_in[16];
    const float* WU2  = (const float*)d_in[17];
    const float* Wg1  = (const float*)d_in[18];
    const float* ag1  = (const float*)d_in[19];
    const float* Wg2  = (const float*)d_in[20];
    const float* ag2  = (const float*)d_in[21];
    float* out = (float*)d_out;

    char* ws = (char*)d_ws;
    size_t o = 0;
    float* buf0 = (float*)(ws + o); o += (size_t)NB * NC * NH * NT * 4;        // y2 / ping
    float* buf1 = (float*)(ws + o); o += (size_t)NB * NC * NH * (NT / 2) * 4;  // pong
    float* sums = (float*)(ws + o); o += (size_t)NB * 2 * NC * NH * 4;
    float* mbuf = (float*)(ws + o); o += (size_t)NB * 2 * 2 * 4;
    float* scl  = (float*)(ws + o); o += (size_t)NB * 4;

    dim3 fgrid(NB * NH, NT / TT);
    front_kernel<<<fgrid, 256, 0, stream>>>(x, Wc1, bc1, bn1g, bn1b, ac1,
                                            Wc2, bc2, bn2g, bn2b, ac2, buf0);

    float* cur = buf0;
    float* nxt = buf1;
    for (int i = 0; i < NL; ++i) {
        int tL = NT >> (i + 1);
        hipMemsetAsync(sums, 0, (size_t)NB * 2 * NC * NH * sizeof(float), stream);
        dim3 grid(NB * NH, (tL + TT - 1) / TT);
        // c = s*even + branchU(s*odd)
        lift_kernel<<<grid, 256, 0, stream>>>(cur, cur, tL,
                                              WU1 + (size_t)i * NC * NC * 3, WU2 + (size_t)i * NC * NC,
                                              aU + i, scl, i > 0, 0, nxt, sums);
        // d = s*odd - branchP(c)   (sums only, d never materialized)
        lift_kernel<<<grid, 256, 0, stream>>>(cur, nxt, tL,
                                              WP1 + (size_t)i * NC * NC * 3, WP2 + (size_t)i * NC * NC,
                                              aP + i, scl, i > 0, 1, nullptr, sums);
        gate_kernel<<<NB, 32, 0, stream>>>(sums, Wg1, Wg2, ag1, ag2, gum, i, tL, mbuf, scl);
        feats_kernel<<<(NB * 2 * NC * NH + 255) / 256, 256, 0, stream>>>(sums, mbuf, i, tL, out);
        float* tmp = cur; cur = nxt; nxt = tmp;
    }
}
